// VariationalGraphEncoder_34497177322134
// MI455X (gfx1250) — compile-verified, matrix operand order checked by reading, not run on hardware
//
#include <hip/hip_runtime.h>
#include <hip/hip_bf16.h>

// MI455X / gfx1250, wave32. bf16 WMMA 16x16x32 with f32 accumulate,
// register-blocked over M (weights stay resident in VGPRs).

typedef __bf16 v16bf __attribute__((ext_vector_type(16)));
typedef float  v8f   __attribute__((ext_vector_type(8)));

#define HDIM 64
#define DIN  128
#define MSUB 4   // M-subtiles (16 rows each) per block

static __device__ __forceinline__ __bf16 f2bf(float f) { return (__bf16)f; }

// ---------------- init: deg = 1 (self loop), agg = 0 ----------------
__global__ void vge_init(float* __restrict__ deg, float* __restrict__ agg, int N) {
    int gid = blockIdx.x * blockDim.x + threadIdx.x;
    if (gid < N) deg[gid] = 1.0f;
    if (gid < N * HDIM) agg[gid] = 0.0f;
}

// ---------------- degree: deg[dst] += 1 over E edges ----------------
__global__ void vge_degree(const long long* __restrict__ ei, float* __restrict__ deg, int E) {
    int e = blockIdx.x * blockDim.x + threadIdx.x;
    if (e < E) {
        int d = (int)ei[(size_t)E + e];
        unsafeAtomicAdd(&deg[d], 1.0f);
    }
}

// ---------------- dinv = rsqrt(deg)  (deg >= 1 always) ----------------
__global__ void vge_dinv(float* __restrict__ deg, int N) {
    int i = blockIdx.x * blockDim.x + threadIdx.x;
    if (i < N) deg[i] = rsqrtf(deg[i]);
}

// ---------------- per-edge norm = dinv[src] * dinv[dst] ----------------
__global__ void vge_norm(const long long* __restrict__ ei, const float* __restrict__ dinv,
                         float* __restrict__ norm, int E) {
    int e = blockIdx.x * blockDim.x + threadIdx.x;
    if (e < E) {
        int s = (int)ei[e];
        int d = (int)ei[(size_t)E + e];
        norm[e] = dinv[s] * dinv[d];
    }
}

// ---------------- GEMM1: h[N,64] = x[N,128] @ W1[64,128]^T  (bf16 WMMA) ---------------
// Block = 128 threads = 4 waves; wave w owns output columns [16w, 16w+16).
// B (the W1 tile, all K=128) is loaded once and held in VGPRs across MSUB M-subtiles.
__global__ void __launch_bounds__(128) vge_gemm1(const float* __restrict__ x,
                                                 const float* __restrict__ W1,
                                                 float* __restrict__ h, int Mtiles) {
    const int wave  = threadIdx.x >> 5;
    const int lane  = threadIdx.x & 31;
    const int m     = lane & 15;        // A-row (M) for this lane, also B/D column (N)
    const int khalf = lane >> 4;        // 0: K 0-7/16-23, 1: K 8-15/24-31

    // Resident B operand: B[k][n] = W1[n][k], 4 K-steps of 32.
    const float* __restrict__ wrow = W1 + ((size_t)wave * 16 + m) * DIN;
    v16bf B[4];
#pragma unroll
    for (int kk = 0; kk < 4; ++kk) {
        const int K0 = kk * 32 + khalf * 8;
#pragma unroll
        for (int i = 0; i < 8; ++i) {
            B[kk][i]     = f2bf(wrow[K0 + i]);
            B[kk][8 + i] = f2bf(wrow[K0 + 16 + i]);
        }
    }

#pragma unroll
    for (int t = 0; t < MSUB; ++t) {
        const int mt = blockIdx.x * MSUB + t;   // wave-uniform
        if (mt >= Mtiles) return;               // uniform exit: EXEC stays all-ones
        const float* __restrict__ xrow = x + ((size_t)mt * 16 + m) * DIN;
        v8f c = {};
#pragma unroll
        for (int kk = 0; kk < 4; ++kk) {        // K = 128 in 4 steps of 32
            const int K0 = kk * 32 + khalf * 8;
            v16bf a;
#pragma unroll
            for (int i = 0; i < 8; ++i) {
                a[i]     = f2bf(xrow[K0 + i]);
                a[8 + i] = f2bf(xrow[K0 + 16 + i]);
            }
            c = __builtin_amdgcn_wmma_f32_16x16x32_bf16(false, a, false, B[kk],
                                                        (short)0, c, false, false);
        }
        // D layout: VGPR r -> M = r + 8*khalf, N = lane&15
        const int mofs = khalf * 8;
#pragma unroll
        for (int r = 0; r < 8; ++r) {
            h[((size_t)mt * 16 + mofs + r) * HDIM + wave * 16 + m] = c[r];
        }
    }
}

// ---------------- message scatter: agg[dst] += h[src] * norm ----------------
// 16 threads per edge, float4 per thread (h rows are 256B aligned).
__global__ void vge_message(const long long* __restrict__ ei, const float* __restrict__ norm,
                            const float4* __restrict__ h4, float* __restrict__ agg, int E) {
    int gid = blockIdx.x * blockDim.x + threadIdx.x;
    if (gid >= E * 16) return;
    const int e = gid >> 4;
    const int q = gid & 15;
    const int s = (int)ei[e];
    const int d = (int)ei[(size_t)E + e];
    const float w = norm[e];
    const float4 hv = h4[(size_t)s * 16 + q];
    float* dst = agg + (size_t)d * HDIM + q * 4;
    unsafeAtomicAdd(dst + 0, hv.x * w);
    unsafeAtomicAdd(dst + 1, hv.y * w);
    unsafeAtomicAdd(dst + 2, hv.z * w);
    unsafeAtomicAdd(dst + 3, hv.w * w);
}

// ------------- finalize: agg = relu(agg + b1 + h*dinv^2)  (self-loop fused) -------------
__global__ void vge_finalize(float* __restrict__ agg, const float* __restrict__ h,
                             const float* __restrict__ dinv, const float* __restrict__ b1,
                             int N) {
    int gid = blockIdx.x * blockDim.x + threadIdx.x;
    if (gid >= N * 16) return;
    const int node = gid >> 4;
    const int q = gid & 15;
    const float dv = dinv[node];
    const float w = dv * dv;
    const size_t base = (size_t)node * HDIM + q * 4;
#pragma unroll
    for (int j = 0; j < 4; ++j) {
        float v = agg[base + j] + b1[q * 4 + j] + h[base + j] * w;
        agg[base + j] = fmaxf(v, 0.0f);
    }
}

// ---------- heads: mu = hr @ Wmu^T + bmu ; lv = hr @ Wlv^T + blv  (bf16 WMMA) ----------
// Both weight tiles stay resident in VGPRs across MSUB M-subtiles.
__global__ void __launch_bounds__(128) vge_heads(const float* __restrict__ hr,
                                                 const float* __restrict__ Wmu,
                                                 const float* __restrict__ bmu,
                                                 const float* __restrict__ Wlv,
                                                 const float* __restrict__ blv,
                                                 float* __restrict__ out, int N, int Mtiles) {
    const int wave  = threadIdx.x >> 5;
    const int lane  = threadIdx.x & 31;
    const int m     = lane & 15;
    const int khalf = lane >> 4;

    const float* __restrict__ wmrow = Wmu + ((size_t)wave * 16 + m) * HDIM;
    const float* __restrict__ wlrow = Wlv + ((size_t)wave * 16 + m) * HDIM;
    v16bf Bm[2], Bl[2];
#pragma unroll
    for (int kk = 0; kk < 2; ++kk) {            // K = 64 in 2 steps of 32
        const int K0 = kk * 32 + khalf * 8;
#pragma unroll
        for (int i = 0; i < 8; ++i) {
            Bm[kk][i]     = f2bf(wmrow[K0 + i]);
            Bm[kk][8 + i] = f2bf(wmrow[K0 + 16 + i]);
            Bl[kk][i]     = f2bf(wlrow[K0 + i]);
            Bl[kk][8 + i] = f2bf(wlrow[K0 + 16 + i]);
        }
    }

    const int ncol = wave * 16 + m;
    const float bmv  = bmu[ncol];
    const float blvv = blv[ncol];
    const size_t NH = (size_t)N * HDIM;
    const int mofs = khalf * 8;

#pragma unroll
    for (int t = 0; t < MSUB; ++t) {
        const int mt = blockIdx.x * MSUB + t;   // wave-uniform
        if (mt >= Mtiles) return;
        const float* __restrict__ arow = hr + ((size_t)mt * 16 + m) * HDIM;
        v8f cmu = {}, clvacc = {};
#pragma unroll
        for (int kk = 0; kk < 2; ++kk) {
            const int K0 = kk * 32 + khalf * 8;
            v16bf a;
#pragma unroll
            for (int i = 0; i < 8; ++i) {
                a[i]     = f2bf(arow[K0 + i]);
                a[8 + i] = f2bf(arow[K0 + 16 + i]);
            }
            cmu    = __builtin_amdgcn_wmma_f32_16x16x32_bf16(false, a, false, Bm[kk],
                                                             (short)0, cmu, false, false);
            clvacc = __builtin_amdgcn_wmma_f32_16x16x32_bf16(false, a, false, Bl[kk],
                                                             (short)0, clvacc, false, false);
        }
#pragma unroll
        for (int r = 0; r < 8; ++r) {
            const size_t row = (size_t)mt * 16 + mofs + r;
            out[row * HDIM + ncol]      = cmu[r] + bmv;      // mu
            out[NH + row * HDIM + ncol] = clvacc[r] + blvv;  // logvar
        }
    }
}

extern "C" void kernel_launch(void* const* d_in, const int* in_sizes, int n_in,
                              void* d_out, int out_size, void* d_ws, size_t ws_size,
                              hipStream_t stream) {
    const float*     x   = (const float*)d_in[0];
    const long long* ei  = (const long long*)d_in[1];
    const float*     W1  = (const float*)d_in[2];
    const float*     b1  = (const float*)d_in[3];
    const float*     Wmu = (const float*)d_in[4];
    const float*     bmu = (const float*)d_in[5];
    const float*     Wlv = (const float*)d_in[6];
    const float*     blv = (const float*)d_in[7];
    float*           out = (float*)d_out;

    const int N = in_sizes[0] / DIN;   // 100000
    const int E = in_sizes[1] / 2;     // 1000000
    const int Mtiles = (N + 15) / 16;  // 6250 (N divisible by 16)

    // Workspace layout (256B aligned slabs)
    char* ws = (char*)d_ws;
    size_t off = 0;
    auto carve = [&](size_t bytes) -> void* {
        void* p = (void*)(ws + off);
        off += (bytes + 255) & ~(size_t)255;
        return p;
    };
    float* dinv = (float*)carve((size_t)N * 4);           // deg, then rsqrt in place
    float* h    = (float*)carve((size_t)N * HDIM * 4);    // pre-aggregation features
    float* agg  = (float*)carve((size_t)N * HDIM * 4);    // aggregated + relu in place
    float* norm = (float*)carve((size_t)E * 4);           // per-edge sym norm

    const int TB = 256;
    const int gemmBlocks = (Mtiles + MSUB - 1) / MSUB;    // 1563
    vge_init    <<<(N * HDIM + TB - 1) / TB, TB, 0, stream>>>(dinv, agg, N);
    vge_degree  <<<(E + TB - 1) / TB, TB, 0, stream>>>(ei, dinv, E);
    vge_dinv    <<<(N + TB - 1) / TB, TB, 0, stream>>>(dinv, N);
    vge_norm    <<<(E + TB - 1) / TB, TB, 0, stream>>>(ei, dinv, norm, E);
    vge_gemm1   <<<gemmBlocks, 128, 0, stream>>>(x, W1, h, Mtiles);
    vge_message <<<((size_t)E * 16 + TB - 1) / TB, TB, 0, stream>>>(ei, norm,
                                                                    (const float4*)h, agg, E);
    vge_finalize<<<(N * 16 + TB - 1) / TB, TB, 0, stream>>>(agg, h, dinv, b1, N);
    vge_heads   <<<gemmBlocks, 128, 0, stream>>>(agg, Wmu, bmu, Wlv, blv, out, N, Mtiles);
}
